// FineMatching_32658931319264
// MI455X (gfx1250) — matrix operand order repaired and checked by reading, not compiled, not run on hardware
//
#include <hip/hip_runtime.h>
#include <hip/hip_bf16.h>

typedef __attribute__((ext_vector_type(16))) _Float16 v16h;
typedef __attribute__((ext_vector_type(8)))  _Float16 v8h;
typedef __attribute__((ext_vector_type(4)))  _Float16 v4h;
typedef __attribute__((ext_vector_type(8)))  float    v8f;

#define WW 100          // W*W
#define CCH 64          // channels
#define TILES 7         // ceil(112/16)
#define PAD_ROWS 112
#define KPAD 128        // padded r-dimension for the moment GEMM
#define BLOCK_T 224     // 7 waves of 32

__global__ __launch_bounds__(BLOCK_T)
void fine_matching_kernel(const float* __restrict__ feat0,
                          const float* __restrict__ feat1,
                          float* __restrict__ outE,   // [M,100,3]
                          float* __restrict__ outCT,  // [M,1,2]
                          float* __restrict__ outDL)  // [M,1,2]
{
    // Staging buffers (f16 features) are dead after the conf GEMM; the
    // per-wave heat tiles H reuse the same LDS behind a barrier.
    __shared__ union {
        struct { _Float16 f0[PAD_ROWS * CCH]; _Float16 f1[PAD_ROWS * CCH]; } st;
        _Float16 h[TILES * 16 * KPAD];   // per-wave H [16 rows][128 r], row-major
    } sU;
    __shared__ _Float16 sG[16 * KPAD];   // G [16 d][128 r]: gx,gy,gx2,gy2,1,0...
    __shared__ float sStd[PAD_ROWS];
    __shared__ float sCx[PAD_ROWS];
    __shared__ float sCy[PAD_ROWS];

    const int m   = blockIdx.x;
    const int tid = threadIdx.x;

    // ---- stage features f32 -> f16, build G, zero pads ----
    {
        const float4* f0v = (const float4*)(feat0 + (size_t)m * (WW * CCH));
        const float4* f1v = (const float4*)(feat1 + (size_t)m * (WW * CCH));
        for (int q = tid; q < (WW * CCH) / 4; q += BLOCK_T) {
            float4 a = f0v[q];
            v4h ha = { (_Float16)a.x, (_Float16)a.y, (_Float16)a.z, (_Float16)a.w };
            *(v4h*)(sU.st.f0 + q * 4) = ha;
            float4 b = f1v[q];
            v4h hb = { (_Float16)b.x, (_Float16)b.y, (_Float16)b.z, (_Float16)b.w };
            *(v4h*)(sU.st.f1 + q * 4) = hb;
        }
        v4h z = {};
        for (int q = tid; q < (PAD_ROWS - WW) * CCH / 4; q += BLOCK_T) {
            *(v4h*)(sU.st.f0 + WW * CCH + q * 4) = z;
            *(v4h*)(sU.st.f1 + WW * CCH + q * 4) = z;
        }
        // G[d][r]: d0=gx, d1=gy, d2=gx^2, d3=gy^2, d4=1, else 0; r>=100 -> 0
        for (int q = tid; q < (16 * KPAD) / 4; q += BLOCK_T) {
            v4h g;
#pragma unroll
            for (int t = 0; t < 4; ++t) {
                const int idx = q * 4 + t;
                const int d = idx >> 7, r = idx & (KPAD - 1);
                float val = 0.0f;
                if (r < WW && d < 5) {
                    const int x = r % 10, y = r / 10;
                    const float gx = -1.0f + (float)x * (2.0f / 9.0f);
                    const float gy = -1.0f + (float)y * (2.0f / 9.0f);
                    val = (d == 0) ? gx : (d == 1) ? gy
                        : (d == 2) ? gx * gx : (d == 3) ? gy * gy : 1.0f;
                }
                g[t] = (_Float16)val;
            }
            *(v4h*)(sG + q * 4) = g;
        }
    }
    __syncthreads();

    const int wave = tid >> 5;       // 0..6 -> conf tile-row
    const int lane = tid & 31;
    const int li   = lane & 15;
    const int hi   = lane >> 4;

    // ---- conf GEMM: tile-row `wave` x 7 column tiles, K=64 ----
    v8f acc[TILES] = {};
#pragma unroll
    for (int k0 = 0; k0 < CCH; k0 += 32) {
        const int abase = (wave * 16 + li) * CCH + k0 + hi * 8;
        v8h alo = *(const v8h*)(sU.st.f0 + abase);
        v8h ahi = *(const v8h*)(sU.st.f0 + abase + 16);
        v16h afrag = __builtin_shufflevector(alo, ahi,
            0,1,2,3,4,5,6,7,8,9,10,11,12,13,14,15);
#pragma unroll
        for (int j = 0; j < TILES; ++j) {
            const int bbase = (j * 16 + li) * CCH + k0 + hi * 8;
            v8h blo = *(const v8h*)(sU.st.f1 + bbase);
            v8h bhi = *(const v8h*)(sU.st.f1 + bbase + 16);
            v16h bfrag = __builtin_shufflevector(blo, bhi,
                0,1,2,3,4,5,6,7,8,9,10,11,12,13,14,15);
            acc[j] = __builtin_amdgcn_wmma_f32_16x16x32_f16(
                false, afrag, false, bfrag, (short)0, acc[j], false, false);
        }
    }
    __syncthreads();   // all waves done reading staged features; LDS now = H

    // ---- e = exp(conf/8) (softmax shift-invariant; |conf| <~ 7 so no max
    //      subtraction needed), stored f16 into this wave's H tile ----
    const int hbase = wave * 16 * KPAD;
    {   // zero pad columns r=112..127 of this wave's H
        v8h z = {};
        *(v8h*)(sU.h + hbase + li * KPAD + PAD_ROWS + hi * 8) = z;
    }
#pragma unroll
    for (int v = 0; v < 8; ++v) {
        const int row = hi * 8 + v;           // local l
#pragma unroll
        for (int j = 0; j < TILES; ++j) {
            const int c = j * 16 + li;        // global r column
            const float e = (c < WW) ? __expf(acc[j][v] * 0.125f) : 0.0f;
            sU.h[hbase + row * KPAD + c] = (_Float16)e;
        }
    }

    // ---- moment GEMM: D[16 l x 16 d] = H[16 x 128] x G^T, 4 WMMAs ----
    v8f mac = {};
#pragma unroll
    for (int k0 = 0; k0 < KPAD; k0 += 32) {
        const int ab = hbase + li * KPAD + k0 + hi * 8;
        v8h alo = *(const v8h*)(sU.h + ab);
        v8h ahi = *(const v8h*)(sU.h + ab + 16);
        v16h afrag = __builtin_shufflevector(alo, ahi,
            0,1,2,3,4,5,6,7,8,9,10,11,12,13,14,15);
        const int bb = li * KPAD + k0 + hi * 8;   // lane li = column d
        v8h blo = *(const v8h*)(sG + bb);
        v8h bhi = *(const v8h*)(sG + bb + 16);
        v16h bfrag = __builtin_shufflevector(blo, bhi,
            0,1,2,3,4,5,6,7,8,9,10,11,12,13,14,15);
        mac = __builtin_amdgcn_wmma_f32_16x16x32_f16(
            false, afrag, false, bfrag, (short)0, mac, false, false);
    }

    // ---- epilogue: lane li holds moment d=li for rows v+8*hi ----
    const float SCALE  = 1.125f;       // (W-1)/RES_RATIO
    const float SCALE2 = 1.265625f;    // SCALE^2
#pragma unroll
    for (int v = 0; v < 8; ++v) {
        const float dv = mac[v];
        const int b  = hi * 16;
        const float px  = __shfl(dv, b + 0, 32);
        const float py  = __shfl(dv, b + 1, 32);
        const float pxx = __shfl(dv, b + 2, 32);
        const float pyy = __shfl(dv, b + 3, 32);
        const float s   = __shfl(dv, b + 4, 32);

        const float inv = 1.0f / s;
        const float cx  = px * inv * SCALE;
        const float cy  = py * inv * SCALE;
        const float vx  = pxx * inv * SCALE2 - cx * cx;
        const float vy  = pyy * inv * SCALE2 - cy * cy;
        const float stdv = sqrtf(fmaxf(vx, 1e-10f)) + sqrtf(fmaxf(vy, 1e-10f));

        const int r = wave * 16 + hi * 8 + v;
        if (li == 0 && r < WW) {
            float* e3 = outE + ((size_t)m * WW + r) * 3;
            e3[0] = cx; e3[1] = cy; e3[2] = stdv;
            sStd[r] = stdv; sCx[r] = cx; sCy[r] = cy;
        }
    }
    __syncthreads();

    // ---- top-1 smallest std (first index wins ties) ----
    if (tid == 0) {
        int best = 0;
        float bs = sStd[0];
        for (int r = 1; r < WW; ++r) {
            const float sr = sStd[r];
            if (sr < bs) { bs = sr; best = r; }
        }
        outCT[(size_t)m * 2 + 0] = sCx[best];
        outCT[(size_t)m * 2 + 1] = sCy[best];
        const int x = best % 10, y = best / 10;
        outDL[(size_t)m * 2 + 0] = (float)x - 4.5f;
        outDL[(size_t)m * 2 + 1] = (float)y - 4.5f;
    }
}

extern "C" void kernel_launch(void* const* d_in, const int* in_sizes, int n_in,
                              void* d_out, int out_size, void* d_ws, size_t ws_size,
                              hipStream_t stream) {
    (void)n_in; (void)out_size; (void)d_ws; (void)ws_size;
    const float* feat0 = (const float*)d_in[0];
    const float* feat1 = (const float*)d_in[1];
    const int M = in_sizes[0] / (WW * CCH);   // 8000

    float* outE  = (float*)d_out;
    float* outCT = outE + (size_t)M * WW * 3;
    float* outDL = outCT + (size_t)M * 2;

    fine_matching_kernel<<<dim3(M), dim3(BLOCK_T), 0, stream>>>(
        feat0, feat1, outE, outCT, outDL);
}
